// UV_Layer_84971632984271
// MI455X (gfx1250) — matrix operand order repaired and checked by reading, not compiled, not run on hardware
//
#include <hip/hip_runtime.h>
#include <hip/hip_bf16.h>

// ---------------------------------------------------------------------------
// Problem constants (reference: T=H=W=128, HID=32, K=3, SAME padding)
// ---------------------------------------------------------------------------
#define T_    128
#define H_    128
#define W_    128
#define HW_   (H_ * W_)          // 16384
#define THW_  (T_ * H_ * W_)     // 2097152
#define HID_  32

typedef __bf16 bf16;
typedef __attribute__((ext_vector_type(16))) bf16  v16bf;
typedef __attribute__((ext_vector_type(8)))  float v8f;

// D = A(16x32 bf16) x B(32x16 bf16) + C(16x16 f32)
__device__ __forceinline__ v8f wmma_bf16(v16bf a, v16bf b, v8f c) {
  return __builtin_amdgcn_wmma_f32_16x16x32_bf16(
      /*neg_a=*/false, a, /*neg_b=*/false, b,
      /*c_mod=*/(short)0, c, /*reuse_a=*/false, /*reuse_b=*/false);
}

// A-fragment K index (contiguous order, matching B):
//   lanes 0-15 : K=e (K 0..15)
//   lanes16-31 : K=e+16 (K 16..31)
__device__ __forceinline__ int a_kidx(int e, int hi) {
  return hi * 16 + e;
}

// CDNA5 async global->LDS copy (ASYNCcnt-tracked, no VGPR round trip).
// LDS address comes from the vdst VGPR; generic LDS pointer's low 32 bits are the
// workgroup-relative LDS byte offset (flat-aperture truncation rule).
__device__ __forceinline__ void async_load_b64(void* lds_dst, const void* gsrc) {
  unsigned lds_off = (unsigned)(size_t)lds_dst;
  asm volatile("global_load_async_to_lds_b64 %0, %1, off"
               :: "v"(lds_off), "v"(gsrc) : "memory");
}
__device__ __forceinline__ void wait_async0() {
  asm volatile("s_wait_asynccnt 0x0" ::: "memory");
}

// ===========================================================================
// K1: Mh = convM (*) M      (1 -> 32 channels), store fp32 + bf16 copies
//   grid = T*H blocks (one (t,h) output row of 128 w), 256 threads = 8 waves,
//   each wave produces 16 w pixels x 32 oc with 2 WMMAs (K=27 padded to 32).
// ===========================================================================
__global__ __launch_bounds__(256) void k_convM(
    const float* __restrict__ M, const float* __restrict__ Wm,
    float* __restrict__ MhF, bf16* __restrict__ MhB)
{
  __shared__ bf16 sIn[9][132];      // (dt,dh) x (w=-1..130), bf16
  __shared__ bf16 sW[2][16][32];    // [ocTile][m][k], k padded 27->32

  const int th = blockIdx.x;
  const int t  = th >> 7, h = th & 127;
  const int tid = threadIdx.x;

  for (int i = tid; i < 2 * 16 * 32; i += 256) {
    int oc = i >> 5, k = i & 31;
    float w = (k < 27) ? Wm[oc * 27 + k] : 0.f;
    sW[oc >> 4][oc & 15][k] = (bf16)w;
  }
  for (int i = tid; i < 9 * 132; i += 256) {
    int r = i / 132, c = i % 132;
    int dt = r / 3, dh = r % 3;
    int tt = t + dt - 1, hh = h + dh - 1, ww = c - 1;
    float v = 0.f;
    if (tt >= 0 && tt < T_ && hh >= 0 && hh < H_ && ww >= 0 && ww < W_)
      v = M[tt * HW_ + hh * W_ + ww];
    sIn[r][c] = (bf16)v;
  }
  __syncthreads();

  const int wave = tid >> 5, lane = tid & 31;
  const int w0 = wave * 16, n = lane & 15, hi = lane >> 4, m = lane & 15;

  v16bf B;
#pragma unroll
  for (int e = 0; e < 16; ++e) {
    int k = hi * 16 + e;             // B: lanes0-15 K0..15, lanes16-31 K16..31
    bf16 v = (bf16)0.f;
    if (k < 27) {
      int dt = k / 9, rr = k % 9, dh = rr / 3, dw = rr % 3;
      v = sIn[dt * 3 + dh][w0 + n + dw];   // lds idx = (w)+1 = w0+n+dw
    }
    B[e] = v;
  }
  v16bf A0, A1;
#pragma unroll
  for (int e = 0; e < 16; ++e) {
    int k = a_kidx(e, hi);
    A0[e] = sW[0][m][k];
    A1[e] = sW[1][m][k];
  }
  v8f c0 = {}, c1 = {};
  c0 = wmma_bf16(A0, B, c0);
  c1 = wmma_bf16(A1, B, c1);

  const long pix = (long)t * HW_ + h * W_ + w0 + n;
#pragma unroll
  for (int i = 0; i < 8; ++i) {
    int oc = i + hi * 8;             // C/D layout: lanes>=16 hold M = i+8
    float v0 = c0[i], v1 = c1[i];
    MhF[(long)oc * THW_ + pix]        = v0;
    MhF[(long)(oc + 16) * THW_ + pix] = v1;
    MhB[(long)oc * THW_ + pix]        = (bf16)v0;
    MhB[(long)(oc + 16) * THW_ + pix] = (bf16)v1;
  }
}

// ===========================================================================
// K2: fused conv0 & conv1 over Mh (32 -> 1 each), one GEMM with M-rows {0,1}.
//   K-chunks: 27 taps, each chunk K=32 channels. LDS restaged per dt slab via
//   CDNA5 async global->LDS (b64 chunks), halos/invalid rows zero-filled.
//   bg2 = (1 - relu(conv0 + b0))^2 ; c1m = 1 - conv1
// ===========================================================================
__global__ __launch_bounds__(256) void k_conv01(
    const bf16* __restrict__ MhB, const float* __restrict__ W0,
    const float* __restrict__ b0, const float* __restrict__ W1,
    float* __restrict__ bg2, float* __restrict__ c1m)
{
  __shared__ alignas(16) bf16 sIn[3][32][136]; // [dh][chan][w+4], ~26 KB
  __shared__ bf16 sW01[2][27][32];             // [row][tap][channel]

  const int th = blockIdx.x;
  const int t  = th >> 7, h = th & 127;
  const int tid = threadIdx.x;

  for (int i = tid; i < 2 * 27 * 32; i += 256) {
    int r = i / (27 * 32), rem = i % (27 * 32);
    int k = rem / 32, c = rem % 32;
    const float* Wp = r ? W1 : W0;
    sW01[r][k][c] = (bf16)Wp[c * 27 + k];
  }

  const int wave = tid >> 5, lane = tid & 31;
  const int w0 = wave * 16, n = lane & 15, hi = lane >> 4, m = lane & 15;

  v8f acc = {};
  for (int dt = 0; dt < 3; ++dt) {
    __syncthreads();
    const int tt = t + dt - 1;
    // halo columns (w=-1 and w=128,129 + padding) are always zero
    for (int r = tid; r < 3 * 32; r += 256) {
      int dh = r >> 5, c = r & 31;
      *(unsigned long long*)&sIn[dh][c][0]   = 0ULL;   // idx 0..3
      *(unsigned long long*)&sIn[dh][c][132] = 0ULL;   // idx 132..135
    }
    // body w=0..127: 32 async b64 chunks per (dh,channel) row
    for (int i = tid; i < 3 * 32 * 32; i += 256) {
      int dh = i >> 10, rem = i & 1023;
      int c = rem >> 5, j = rem & 31;
      int hh = h + dh - 1;
      bf16* dst = &sIn[dh][c][4 + j * 4];
      if (tt >= 0 && tt < T_ && hh >= 0 && hh < H_) {
        const bf16* src = &MhB[(long)c * THW_ + (long)tt * HW_ + hh * W_ + j * 4];
        async_load_b64(dst, src);
      } else {
        *(unsigned long long*)dst = 0ULL;
      }
    }
    wait_async0();          // drain this wave's ASYNCcnt before the barrier
    __syncthreads();

    for (int j = 0; j < 9; ++j) {
      int dh = j / 3, dw = j % 3, tap = dt * 9 + j;
      v16bf A, B;
#pragma unroll
      for (int e = 0; e < 16; ++e) {
        int kc = hi * 16 + e;                       // K axis == channel
        B[e] = sIn[dh][kc][w0 + n + dw + 3];        // idx = (w)+4
        int ka = a_kidx(e, hi);
        A[e] = (m < 2) ? sW01[m][tap][ka] : (bf16)0.f;
      }
      acc = wmma_bf16(A, B, acc);
    }
  }

  if (hi == 0) {                     // lanes 0-15: acc[0]=M-row0, acc[1]=M-row1
    const long pix = (long)t * HW_ + h * W_ + w0 + n;
    float v0 = acc[0] + b0[0];
    float bg = 1.f - fmaxf(v0, 0.f);
    bg2[pix] = bg * bg;
    c1m[pix] = 1.f - acc[1];
  }
}

// ===========================================================================
// K3: rank-1 U/V fit (bandwidth-bound VALU). Raw-view semantics:
//   temp[t,h,w] = bg2*(X - U[t*128+h]*V[w])
//   gradU[h*128+w] = sum_t temp*V[t]           (old U, old V)
//   gradV[t]       = sum_hw temp*U_new[h*128+w]
// ===========================================================================
__global__ void k_copyUV(const float* __restrict__ U, const float* __restrict__ V,
                         float* __restrict__ Ua, float* __restrict__ Va) {
  int i = blockIdx.x * blockDim.x + threadIdx.x;
  if (i < HW_) Ua[i] = U[i];
  if (i < T_)  Va[i] = V[i];
}

__global__ __launch_bounds__(256) void k_updU(
    const float* __restrict__ X, const float* __restrict__ bg2,
    const float* __restrict__ Uold, const float* __restrict__ Vold,
    const float* __restrict__ ulearn, int it, float* __restrict__ Unew)
{
  const int hw = blockIdx.x * 256 + threadIdx.x;   // 64 blocks x 256 = HW_
  const int h = hw >> 7, w = hw & 127;
  const float Vw = Vold[w];
  float g = 0.f;
  for (int t = 0; t < T_; ++t) {
    if (t + 1 < T_) __builtin_prefetch(&X[(long)(t + 1) * HW_ + hw], 0, 1);
    float u_th = Uold[t * 128 + h];
    float temp = bg2[(long)t * HW_ + hw] * (X[(long)t * HW_ + hw] - u_th * Vw);
    g += temp * Vold[t];
  }
  Unew[hw] = Uold[hw] + 1e-5f * ulearn[it] * g;
}

__global__ __launch_bounds__(256) void k_updV(
    const float* __restrict__ X, const float* __restrict__ bg2,
    const float* __restrict__ Uold, const float* __restrict__ Unew,
    const float* __restrict__ Vold, const float* __restrict__ vlearn,
    int it, float* __restrict__ Vnew)
{
  __shared__ float red[256];
  const int t = blockIdx.x;
  float s = 0.f;
  for (int hw = threadIdx.x; hw < HW_; hw += 256) {
    int h = hw >> 7, w = hw & 127;
    float temp = bg2[(long)t * HW_ + hw] *
                 (X[(long)t * HW_ + hw] - Uold[t * 128 + h] * Vold[w]);
    s += temp * Unew[hw];
  }
  red[threadIdx.x] = s;
  __syncthreads();
  for (int off = 128; off > 0; off >>= 1) {
    if (threadIdx.x < (unsigned)off) red[threadIdx.x] += red[threadIdx.x + off];
    __syncthreads();
  }
  if (threadIdx.x == 0) Vnew[t] = Vold[t] + 1e-5f * vlearn[it] * red[0];
}

// ===========================================================================
// K4: s = c1m * (X - U[h*128+w]*V[t])^2 generated on the fly (fg2 fused),
//     _X = tau_s * (conv2 (*) s)  (1 -> 32 ch, same GEMM as K1),
//     Mtmp = Mh + _X, soft-threshold with th[w]*g2[h]*g[oc], store bf16.
// ===========================================================================
__global__ __launch_bounds__(256) void k_conv2(
    const float* __restrict__ X, const float* __restrict__ c1m,
    const float* __restrict__ Uf, const float* __restrict__ Vf,
    const float* __restrict__ W2, const float* __restrict__ tau_raw,
    const float* __restrict__ MhF, const float* __restrict__ g,
    const float* __restrict__ th_w, const float* __restrict__ g2,
    bf16* __restrict__ MtB)
{
  __shared__ bf16 sIn[9][132];
  __shared__ bf16 sW[2][16][32];

  const int th = blockIdx.x;
  const int t  = th >> 7, h = th & 127;
  const int tid = threadIdx.x;

  for (int i = tid; i < 2 * 16 * 32; i += 256) {
    int oc = i >> 5, k = i & 31;
    float w = (k < 27) ? W2[oc * 27 + k] : 0.f;
    sW[oc >> 4][oc & 15][k] = (bf16)w;
  }
  for (int i = tid; i < 9 * 132; i += 256) {
    int r = i / 132, c = i % 132;
    int dt = r / 3, dh = r % 3;
    int tt = t + dt - 1, hh = h + dh - 1, ww = c - 1;
    float v = 0.f;
    if (tt >= 0 && tt < T_ && hh >= 0 && hh < H_ && ww >= 0 && ww < W_) {
      long p = (long)tt * HW_ + hh * W_ + ww;
      float L = Uf[hh * W_ + ww] * Vf[tt];       // L[t,h,w] = U[hw]*V[t]
      float d = X[p] - L;
      v = c1m[p] * d * d;
    }
    sIn[r][c] = (bf16)v;
  }
  __syncthreads();

  const int wave = tid >> 5, lane = tid & 31;
  const int w0 = wave * 16, n = lane & 15, hi = lane >> 4, m = lane & 15;

  v16bf B;
#pragma unroll
  for (int e = 0; e < 16; ++e) {
    int k = hi * 16 + e;
    bf16 v = (bf16)0.f;
    if (k < 27) {
      int dt = k / 9, rr = k % 9, dh = rr / 3, dw = rr % 3;
      v = sIn[dt * 3 + dh][w0 + n + dw];
    }
    B[e] = v;
  }
  v16bf A0, A1;
#pragma unroll
  for (int e = 0; e < 16; ++e) {
    int k = a_kidx(e, hi);
    A0[e] = sW[0][m][k];
    A1[e] = sW[1][m][k];
  }
  v8f c0 = {}, c1 = {};
  c0 = wmma_bf16(A0, B, c0);
  c1 = wmma_bf16(A1, B, c1);

  const float tau_s = logf(1.f + expf(tau_raw[0]));   // softplus
  const int ww = w0 + n;
  const long pix = (long)t * HW_ + h * W_ + ww;
  const float thg = th_w[ww] * g2[h];
#pragma unroll
  for (int i = 0; i < 8; ++i) {
    int oc0 = i + hi * 8;
    int oc1 = oc0 + 16;
    float v0 = MhF[(long)oc0 * THW_ + pix] + tau_s * c0[i];
    float v1 = MhF[(long)oc1 * THW_ + pix] + tau_s * c1[i];
    float th0 = thg * g[oc0], th1 = thg * g[oc1];
    float z0 = copysignf(fmaxf(fabsf(v0) - th0, 0.f), v0);
    float z1 = copysignf(fmaxf(fabsf(v1) - th1, 0.f), v1);
    MtB[(long)oc0 * THW_ + pix] = (bf16)z0;
    MtB[(long)oc1 * THW_ + pix] = (bf16)z1;
  }
}

// ===========================================================================
// K5: conv4 (*) Mt (32 -> 1) + sigmoid((y - 0.5)*mask_scaling) -> M_out
//   Same async-staged 32-channel GEMM as K2, 1 useful M row.
// ===========================================================================
__global__ __launch_bounds__(256) void k_conv4(
    const bf16* __restrict__ MtB, const float* __restrict__ W4,
    const float* __restrict__ mscale, float* __restrict__ Mout)
{
  __shared__ alignas(16) bf16 sIn[3][32][136];
  __shared__ bf16 sW4[27][32];

  const int th = blockIdx.x;
  const int t  = th >> 7, h = th & 127;
  const int tid = threadIdx.x;

  for (int i = tid; i < 27 * 32; i += 256) {
    int k = i / 32, c = i % 32;
    sW4[k][c] = (bf16)W4[c * 27 + k];
  }

  const int wave = tid >> 5, lane = tid & 31;
  const int w0 = wave * 16, n = lane & 15, hi = lane >> 4, m = lane & 15;

  v8f acc = {};
  for (int dt = 0; dt < 3; ++dt) {
    __syncthreads();
    const int tt = t + dt - 1;
    for (int r = tid; r < 3 * 32; r += 256) {
      int dh = r >> 5, c = r & 31;
      *(unsigned long long*)&sIn[dh][c][0]   = 0ULL;
      *(unsigned long long*)&sIn[dh][c][132] = 0ULL;
    }
    for (int i = tid; i < 3 * 32 * 32; i += 256) {
      int dh = i >> 10, rem = i & 1023;
      int c = rem >> 5, j = rem & 31;
      int hh = h + dh - 1;
      bf16* dst = &sIn[dh][c][4 + j * 4];
      if (tt >= 0 && tt < T_ && hh >= 0 && hh < H_) {
        const bf16* src = &MtB[(long)c * THW_ + (long)tt * HW_ + hh * W_ + j * 4];
        async_load_b64(dst, src);
      } else {
        *(unsigned long long*)dst = 0ULL;
      }
    }
    wait_async0();
    __syncthreads();

    for (int j = 0; j < 9; ++j) {
      int dh = j / 3, dw = j % 3, tap = dt * 9 + j;
      v16bf A, B;
#pragma unroll
      for (int e = 0; e < 16; ++e) {
        int kc = hi * 16 + e;
        B[e] = sIn[dh][kc][w0 + n + dw + 3];
        int ka = a_kidx(e, hi);
        A[e] = (m == 0) ? sW4[tap][ka] : (bf16)0.f;
      }
      acc = wmma_bf16(A, B, acc);
    }
  }

  if (hi == 0) {                                   // M-row 0 lives in lanes 0-15
    const long pix = (long)t * HW_ + h * W_ + w0 + n;
    float y = (acc[0] - 0.5f) * mscale[0];
    Mout[pix] = 1.f / (1.f + expf(-y));
  }
}

// ===========================================================================
// Host side
// ===========================================================================
extern "C" void kernel_launch(void* const* d_in, const int* in_sizes, int n_in,
                              void* d_out, int out_size, void* d_ws, size_t ws_size,
                              hipStream_t stream) {
  (void)in_sizes; (void)n_in; (void)out_size; (void)ws_size;

  const float* X      = (const float*)d_in[0];
  const float* U      = (const float*)d_in[1];
  const float* V      = (const float*)d_in[2];
  const float* M      = (const float*)d_in[3];
  const float* convMw = (const float*)d_in[4];
  const float* conv0w = (const float*)d_in[5];
  const float* conv0b = (const float*)d_in[6];
  const float* conv1w = (const float*)d_in[7];
  const float* conv2w = (const float*)d_in[8];
  const float* conv4w = (const float*)d_in[9];
  const float* tau_raw  = (const float*)d_in[10];
  const float* mscale   = (const float*)d_in[11];
  const float* g        = (const float*)d_in[12];
  const float* th       = (const float*)d_in[13];
  const float* g2       = (const float*)d_in[14];
  const float* ulearn   = (const float*)d_in[15];
  const float* vlearn   = (const float*)d_in[16];

  // Workspace layout (~554 MB; MI455X has 432 GiB, harness ws assumed ample)
  char* ws = (char*)d_ws;
  size_t off = 0;
  float* MhF = (float*)(ws + off); off += (size_t)HID_ * THW_ * 4;  // 268 MB
  bf16*  MhB = (bf16*) (ws + off); off += (size_t)HID_ * THW_ * 2;  // 134 MB
  bf16*  MtB = (bf16*) (ws + off); off += (size_t)HID_ * THW_ * 2;  // 134 MB
  float* bg2 = (float*)(ws + off); off += (size_t)THW_ * 4;
  float* c1m = (float*)(ws + off); off += (size_t)THW_ * 4;
  float* UA  = (float*)(ws + off); off += HW_ * 4;
  float* UB  = (float*)(ws + off); off += HW_ * 4;
  float* VA  = (float*)(ws + off); off += 512;
  float* VB  = (float*)(ws + off); off += 512;

  const dim3 rows(T_ * H_);   // 16384 blocks: one (t,h) row each
  const dim3 blk(256);

  // 1) Mh = convM (*) M   [WMMA]
  k_convM<<<rows, blk, 0, stream>>>(M, convMw, MhF, MhB);

  // 2) bg2 / c1m from fused conv0+conv1   [WMMA + async->LDS]
  k_conv01<<<rows, blk, 0, stream>>>(MhB, conv0w, conv0b, conv1w, bg2, c1m);

  // 3) 5-step U/V fit (ping-pong buffers)
  k_copyUV<<<64, 256, 0, stream>>>(U, V, UA, VA);
  float *Ucur = UA, *Unxt = UB, *Vcur = VA, *Vnxt = VB;
  for (int i = 0; i < 5; ++i) {
    k_updU<<<HW_ / 256, 256, 0, stream>>>(X, bg2, Ucur, Vcur, ulearn, i, Unxt);
    k_updV<<<T_, 256, 0, stream>>>(X, bg2, Ucur, Unxt, Vcur, vlearn, i, Vnxt);
    float* tu = Ucur; Ucur = Unxt; Unxt = tu;
    float* tv = Vcur; Vcur = Vnxt; Vnxt = tv;
  }

  // 4) conv2 with fused fg2 generation and soft-threshold epilogue   [WMMA]
  k_conv2<<<rows, blk, 0, stream>>>(X, c1m, Ucur, Vcur, conv2w, tau_raw,
                                    MhF, g, th, g2, MtB);

  // 5) conv4 + sigmoid -> M_out section of d_out   [WMMA + async->LDS]
  float* out  = (float*)d_out;
  float* Mout = out + THW_ + HW_ + T_;
  k_conv4<<<rows, blk, 0, stream>>>(MtB, conv4w, mscale, Mout);

  // 6) Pass-through outputs: X, U, V
  (void)hipMemcpyAsync(out,               X,    (size_t)THW_ * 4, hipMemcpyDeviceToDevice, stream);
  (void)hipMemcpyAsync(out + THW_,        Ucur, (size_t)HW_ * 4,  hipMemcpyDeviceToDevice, stream);
  (void)hipMemcpyAsync(out + THW_ + HW_,  Vcur, (size_t)T_ * 4,   hipMemcpyDeviceToDevice, stream);
}
